// MultiHeadAttention_35897336660262
// MI455X (gfx1250) — matrix operand order repaired and checked by reading, not compiled, not run on hardware
//
#include <hip/hip_runtime.h>

#define Lseq 8192
#define Edim 1024
#define Hn   16
#define Dh   64
#define SCALE 0.125f   // 1/sqrt(64)

typedef __attribute__((ext_vector_type(16))) __bf16 v16bf;
typedef __attribute__((ext_vector_type(8)))  float  v8f;

__device__ __forceinline__ __bf16 tobf(float f) { return (__bf16)f; }

// ---------------- small scalar kernels ----------------

// q[e] = x[0,:] . Wq[e,:] + bq[e]
__global__ void mha_qproj(const float* __restrict__ x, const float* __restrict__ ipw,
                          const float* __restrict__ ipb, float* __restrict__ q) {
  int e = blockIdx.x * blockDim.x + threadIdx.x;
  if (e >= Edim) return;
  const float* w = ipw + (size_t)e * Edim;
  float acc = ipb[e];
  for (int j = 0; j < Edim; ++j) acc += x[j] * w[j];
  q[e] = acc;
}

// wqk[h,j] = sum_d q[h*64+d] * Wk[h*64+d, j] ;  sbias[h] = sum_d q[h*64+d]*bk[h*64+d]
__global__ void mha_fold(const float* __restrict__ q, const float* __restrict__ ipw,
                         const float* __restrict__ ipb, float* __restrict__ wqk,
                         float* __restrict__ sbias) {
  const float* Wk = ipw + (size_t)Edim * Edim;
  const float* bk = ipb + Edim;
  int idx = blockIdx.x * blockDim.x + threadIdx.x;      // 0..16383
  int h = idx >> 10, j = idx & 1023;
  float acc = 0.f;
  for (int d = 0; d < Dh; ++d) acc += q[h * Dh + d] * Wk[(size_t)(h * Dh + d) * Edim + j];
  wqk[idx] = acc;
  if (j == 0) {
    float sb = 0.f;
    for (int d = 0; d < Dh; ++d) sb += q[h * Dh + d] * bk[h * Dh + d];
    sbias[h] = sb;
  }
}

// ---------------- WMMA kernel 1: scores = x @ wqk^T  [8192,1024]x[1024,16] ----------------
// 8 waves/block, each wave owns a 16-row l-tile; grid covers L/128 = 64 blocks.
__global__ void mha_scores(const float* __restrict__ x, const float* __restrict__ wqk,
                           const float* __restrict__ sbias, float* __restrict__ s) {
  const int tid  = threadIdx.x;
  const int wave = tid >> 5;
  const int lane = tid & 31;
  const int hi   = lane >> 4;
  const int l0   = blockIdx.x * 128 + wave * 16;

  const int arow = l0 + (lane & 15);          // A: M = lane&15
  const int kbA  = hi ? 8 : 0;                // lanes16-31 hold K 8-15 / 24-31
  const int brow = lane & 15;                 // B: N = lane&15
  const int kbB  = hi ? 8 : 0;                // lanes16-31 hold K 8-15 / 24-31

  const float* __restrict__ ap = x   + (size_t)arow * Edim + kbA;
  const float* __restrict__ bp = wqk + (size_t)brow * Edim + kbB;

  v8f acc = {};
  for (int k0 = 0; k0 < Edim; k0 += 32) {
    v16bf a, b;
#pragma unroll
    for (int i = 0; i < 8; ++i) {
      a[i]     = tobf(ap[k0 + i]);            // K = kbA + 0..7
      a[i + 8] = tobf(ap[k0 + 16 + i]);       // K = kbA + 16..23
    }
#pragma unroll
    for (int i = 0; i < 16; ++i) b[i] = tobf(bp[k0 + (i < 8 ? i : i + 8)]);
    acc = __builtin_amdgcn_wmma_f32_16x16x32_bf16(false, a, false, b, (short)0, acc,
                                                  false, false);
  }
  const int h  = lane & 15;
  const float sb = sbias[h];
#pragma unroll
  for (int r = 0; r < 8; ++r) {
    int l = l0 + r + (hi ? 8 : 0);            // C: M = r (+8 for hi lanes)
    s[(size_t)l * Hn + h] = (acc[r] + sb) * SCALE;
  }
}

// ---------------- softmax per head, writes transposed attn [H][L] ----------------
__global__ void mha_softmax(const float* __restrict__ s, float* __restrict__ attnT) {
  __shared__ float red[256];
  const int h = blockIdx.x, tid = threadIdx.x;
  float m = -1e30f;
  for (int l = tid; l < Lseq; l += 256) m = fmaxf(m, s[(size_t)l * Hn + h]);
  red[tid] = m; __syncthreads();
  for (int o = 128; o > 0; o >>= 1) { if (tid < o) red[tid] = fmaxf(red[tid], red[tid + o]); __syncthreads(); }
  m = red[0]; __syncthreads();
  float sum = 0.f;
  for (int l = tid; l < Lseq; l += 256) sum += __expf(s[(size_t)l * Hn + h] - m);
  red[tid] = sum; __syncthreads();
  for (int o = 128; o > 0; o >>= 1) { if (tid < o) red[tid] += red[tid + o]; __syncthreads(); }
  const float inv = 1.0f / red[0];
  for (int l = tid; l < Lseq; l += 256)
    attnT[(size_t)h * Lseq + l] = __expf(s[(size_t)l * Hn + h] - m) * inv;
}

// ---------------- WMMA kernel 2: y = attn @ x  [16,8192]x[8192,1024] ----------------
// grid = 64 N-tiles; 8 waves split K=8192, LDS reduction combines them.
__global__ void mha_ctxgemm(const float* __restrict__ attnT, const float* __restrict__ x,
                            float* __restrict__ y) {
  __shared__ float red[8][32][8];
  const int tid  = threadIdx.x;
  const int wave = tid >> 5;
  const int lane = tid & 31;
  const int hi   = lane >> 4;
  const int n0   = blockIdx.x * 16;
  const int kstart = wave * (Lseq / 8);

  const int m   = lane & 15;                  // A row = head
  const int kbA = hi ? 8 : 0;
  const int n   = lane & 15;                  // B col
  const int kbB = hi ? 8 : 0;

  const float* __restrict__ ap = attnT + (size_t)m * Lseq + kstart + kbA;
  const float* __restrict__ bp = x + (size_t)(kstart + kbB) * Edim + n0 + n;

  v8f acc = {};
  for (int k0 = 0; k0 < Lseq / 8; k0 += 32) {
    v16bf a, b;
#pragma unroll
    for (int i = 0; i < 8; ++i) {
      a[i]     = tobf(ap[k0 + i]);
      a[i + 8] = tobf(ap[k0 + 16 + i]);
    }
#pragma unroll
    for (int i = 0; i < 16; ++i) b[i] = tobf(bp[(size_t)(k0 + (i < 8 ? i : i + 8)) * Edim]);
    acc = __builtin_amdgcn_wmma_f32_16x16x32_bf16(false, a, false, b, (short)0, acc,
                                                  false, false);
  }
#pragma unroll
  for (int r = 0; r < 8; ++r) red[wave][lane][r] = acc[r];
  __syncthreads();
  {
    int rl = tid & 31, rr = tid >> 5;         // one thread per (lane, vgpr-row)
    float sum = 0.f;
#pragma unroll
    for (int w = 0; w < 8; ++w) sum += red[w][rl][rr];
    int hh = rr + ((rl >= 16) ? 8 : 0);       // C: M = r (+8 for hi lanes)
    int nn = n0 + (rl & 15);
    y[(size_t)hh * Edim + nn] = sum;
  }
}

// ctx[e] = y[h(e),:] . Wv[e,:] + bv[e]
__global__ void mha_ctx(const float* __restrict__ y, const float* __restrict__ ipw,
                        const float* __restrict__ ipb, float* __restrict__ ctx) {
  const float* Wv = ipw + (size_t)2 * Edim * Edim;
  const float* bv = ipb + 2 * Edim;
  int e = blockIdx.x * blockDim.x + threadIdx.x;
  if (e >= Edim) return;
  const float* yr = y + (size_t)(e >> 6) * Edim;
  const float* w  = Wv + (size_t)e * Edim;
  float acc = bv[e];
  for (int j = 0; j < Edim; ++j) acc += yr[j] * w[j];
  ctx[e] = acc;
}

// out[e] = ctx . Wo[e,:] + bo[e]
__global__ void mha_outproj(const float* __restrict__ ctx, const float* __restrict__ opw,
                            const float* __restrict__ opb, float* __restrict__ out) {
  int e = blockIdx.x * blockDim.x + threadIdx.x;
  if (e >= Edim) return;
  const float* w = opw + (size_t)e * Edim;
  float acc = opb[e];
  for (int j = 0; j < Edim; ++j) acc += ctx[j] * w[j];
  out[e] = acc;
}

extern "C" void kernel_launch(void* const* d_in, const int* in_sizes, int n_in,
                              void* d_out, int out_size, void* d_ws, size_t ws_size,
                              hipStream_t stream) {
  const float* x   = (const float*)d_in[0];
  const float* ipw = (const float*)d_in[1];
  const float* ipb = (const float*)d_in[2];
  const float* opw = (const float*)d_in[3];
  const float* opb = (const float*)d_in[4];
  float* out = (float*)d_out;
  float* ws  = (float*)d_ws;

  float* q     = ws;                 // 1024
  float* wqk   = ws + 1024;          // 16*1024
  float* sbias = ws + 17408;         // 16
  float* s     = ws + 17424;         // 8192*16
  float* attnT = ws + 148496;        // 16*8192
  float* y     = ws + 279568;        // 16*1024
  float* ctx   = ws + 295952;        // 1024

  mha_qproj  <<<4,  256, 0, stream>>>(x, ipw, ipb, q);
  mha_fold   <<<64, 256, 0, stream>>>(q, ipw, ipb, wqk, sbias);
  mha_scores <<<64, 256, 0, stream>>>(x, wqk, sbias, s);
  mha_softmax<<<16, 256, 0, stream>>>(s, attnT);
  mha_ctxgemm<<<64, 256, 0, stream>>>(attnT, x, y);
  mha_ctx    <<<4,  256, 0, stream>>>(y, ipw, ipb, ctx);
  mha_outproj<<<4,  256, 0, stream>>>(ctx, opw, opb, out);
}
